// Pointnet_Tracking_62285615726632
// MI455X (gfx1250) — compile-verified, matrix operand order checked by reading, not run on hardware
//
// MI455X (gfx1250) implementation of the PointNet++ tracking pipeline.
// Design: grouped SA MLPs fused (gather -> 3x WMMA GEMM in LDS -> maxpool) so the
// huge grouped tensors never touch HBM; all conv1d chains are f16-in/f32-acc
// v_wmma_f32_16x16x32_f16 GEMMs over point-major [B*N, C] activations.
// Param flatten order assumed = jax pytree (sorted dict keys, 'W' < 'b').

#include <hip/hip_runtime.h>
#include <hip/hip_bf16.h>
#include <cstdint>
#include <cstddef>

#define BATCH 64

typedef _Float16 h16 __attribute__((ext_vector_type(16)));
typedef _Float16 h8v __attribute__((ext_vector_type(8)));
typedef float    f8v __attribute__((ext_vector_type(8)));

// ---------------------------------------------------------------- WMMA helpers
// A-fragment (16x32 f16) per CDNA5 wave32 layout:
//   lanes 0-15  : row M=lane,  K = {k0..k0+7,  k0+16..k0+23}
//   lanes 16-31 : row M=lane-16, K = {k0+8..k0+15, k0+24..k0+31}
// The same loader serves the B-fragment with "row" = output column (N), since
// B[k][n] = W[n][k] and W is stored row-major [co_pad, ci_pad].
static __device__ __forceinline__ h16 load_frag(const _Float16* __restrict__ base,
                                                int ld, int r0, int k0, int lane) {
  int row = r0 + (lane & 15);
  int ko  = k0 + ((lane >> 4) << 3);
  const _Float16* p = base + (size_t)row * ld + ko;
  h8v lo = *(const h8v*)(p);
  h8v hi = *(const h8v*)(p + 16);
  h16 a;
#pragma unroll
  for (int i = 0; i < 8; ++i) { a[i] = lo[i]; a[8 + i] = hi[i]; }
  return a;
}

// -------------------------------------------------------------- generic GEMM
// out[M, co] = act(A[M,Kp] * W^T + bias).  ACT: 0=none 1=relu 2=sigmoid.
template <int ACT, int OUT16>
__global__ __launch_bounds__(256) void k_gemm(const _Float16* __restrict__ A, int lda,
                                              const _Float16* __restrict__ W, int ldw,
                                              const float* __restrict__ bias,
                                              void* __restrict__ out, int ldo,
                                              int tiles_m, int Kp) {
  int lane = threadIdx.x & 31;
  int wid  = threadIdx.x >> 5;
  int mt   = blockIdx.x * (blockDim.x >> 5) + wid;   // wave-uniform
  if (mt >= tiles_m) return;                          // wave-uniform exit
  int m0 = mt * 16;
  int n0 = blockIdx.y * 16;
  f8v acc = {};
  for (int k0 = 0; k0 < Kp; k0 += 32) {
    h16 av = load_frag(A, lda, m0, k0, lane);
    h16 bv = load_frag(W, ldw, n0, k0, lane);
    acc = __builtin_amdgcn_wmma_f32_16x16x32_f16(false, av, false, bv,
                                                 (short)0, acc, false, false);
  }
  int col   = n0 + (lane & 15);
  float bv  = bias[col];
  int rbase = m0 + ((lane >> 4) << 3);
#pragma unroll
  for (int v = 0; v < 8; ++v) {
    float x = acc[v] + bv;
    if (ACT == 1) x = fmaxf(x, 0.0f);
    if (ACT == 2) x = 1.0f / (1.0f + __expf(-x));
    int row = rbase + v;
    if (OUT16) ((_Float16*)out)[(size_t)row * ldo + col] = (_Float16)x;
    else       ((float*)out)[(size_t)row * ldo + col]    = x;
  }
}

// -------------------------------------------------- fused SA grouped MLP + max
struct SAParams {
  const float* xyz;    // [B,N,3]
  const float* feats;  // [B,N,FS] or null
  const int*   idx;    // [B,S,ns]
  const float* cen;    // [B,S,3]
  int N, S, ns, Cf, FS;
  const _Float16* Wl[3];
  const float*    bl[3];
  int K0;        // padded Cin (mult 32)
  int Co[3];     // per-layer out channels
  int nlayers;
  float* out;    // [B,S,OS]
  int OS;
};

__global__ __launch_bounds__(128) void k_sa(SAParams p) {
  __shared__ __align__(16) _Float16 buf0[48 * 288];
  __shared__ __align__(16) _Float16 buf1[48 * 288];
  int b = blockIdx.y, s = blockIdx.x;
  int tid = threadIdx.x;
  const int*   gi  = p.idx + ((size_t)b * p.S + s) * p.ns;
  const float* cen = p.cen + ((size_t)b * p.S + s) * 3;
  int Cin = 3 + p.Cf;
  // gather grouped inputs -> f16 LDS tile [ns, K0] (zero padded)
  for (int t = tid; t < p.ns * p.K0; t += blockDim.x) {
    int r = t / p.K0, c = t - r * p.K0;
    float v = 0.0f;
    int j = gi[r];
    if (c < 3)        v = p.xyz[((size_t)b * p.N + j) * 3 + c] - cen[c];
    else if (c < Cin) v = p.feats[((size_t)b * p.N + j) * p.FS + (c - 3)];
    buf0[t] = (_Float16)v;
  }
  __syncthreads();
  int lane = tid & 31, wid = tid >> 5, nw = blockDim.x >> 5;
  int K = p.K0;
  for (int l = 0; l < p.nlayers; ++l) {
    const _Float16* src = (l & 1) ? buf1 : buf0;
    _Float16*       dst = (l & 1) ? buf0 : buf1;
    int Co = p.Co[l];
    int mt = p.ns >> 4, nt = Co >> 4;
    for (int t = wid; t < mt * nt; t += nw) {   // wave-uniform
      int mi = t % mt, ni = t / mt;
      f8v acc = {};
      for (int k0 = 0; k0 < K; k0 += 32) {
        h16 av = load_frag(src, K, mi * 16, k0, lane);
        h16 bv = load_frag(p.Wl[l], K, ni * 16, k0, lane);
        acc = __builtin_amdgcn_wmma_f32_16x16x32_f16(false, av, false, bv,
                                                     (short)0, acc, false, false);
      }
      int col   = ni * 16 + (lane & 15);
      float bvs = p.bl[l][col];
      int rbase = mi * 16 + ((lane >> 4) << 3);
#pragma unroll
      for (int v = 0; v < 8; ++v) {
        float x = fmaxf(acc[v] + bvs, 0.0f);    // SA layers are all ReLU
        dst[(size_t)(rbase + v) * Co + col] = (_Float16)x;
      }
    }
    __syncthreads();
    K = Co;
  }
  const _Float16* fin = (p.nlayers & 1) ? buf1 : buf0;
  int C = p.Co[p.nlayers - 1];
  for (int c = tid; c < C; c += blockDim.x) {
    float m = -3.0e38f;
    for (int r = 0; r < p.ns; ++r) m = fmaxf(m, (float)fin[r * C + c]);
    p.out[((size_t)b * p.S + s) * p.OS + c] = m;
  }
}

// ------------------------------------------------------------- FPS (argmax)
__global__ __launch_bounds__(256) void k_fps(const float* __restrict__ xyz,
                                             int N, int S, int* __restrict__ idx) {
  __shared__ float dist[1024];
  __shared__ float rv[256];
  __shared__ int   ri[256];
  __shared__ int   s_last;
  int b = blockIdx.x, tid = threadIdx.x;
  const float* base = xyz + (size_t)b * N * 3;
  for (int j = tid; j < N; j += 256) dist[j] = 1.0e10f;
  if (tid == 0) { idx[(size_t)b * S] = 0; s_last = 0; }
  __syncthreads();
  for (int it = 1; it < S; ++it) {
    int last = s_last;
    float lx = base[last * 3], ly = base[last * 3 + 1], lz = base[last * 3 + 2];
    float bv = -1.0f; int bi = 0;
    for (int j = tid; j < N; j += 256) {
      float dx = base[j * 3] - lx, dy = base[j * 3 + 1] - ly, dz = base[j * 3 + 2] - lz;
      float d = fminf(dist[j], dx * dx + dy * dy + dz * dz);
      dist[j] = d;
      if (d > bv) { bv = d; bi = j; }
    }
    rv[tid] = bv; ri[tid] = bi;
    __syncthreads();
    for (int off = 128; off > 0; off >>= 1) {
      if (tid < off) {
        float v2 = rv[tid + off]; int i2 = ri[tid + off];
        if (v2 > rv[tid] || (v2 == rv[tid] && i2 < ri[tid])) { rv[tid] = v2; ri[tid] = i2; }
      }
      __syncthreads();
    }
    if (tid == 0) { s_last = ri[0]; idx[(size_t)b * S + it] = ri[0]; }
    __syncthreads();
  }
}

// ------------------------------------------------------------ small kernels
__global__ void k_gather3(const float* xyz, const int* idx, float* out, int N, int S, int total) {
  int i = blockIdx.x * blockDim.x + threadIdx.x;
  if (i >= total) return;
  int c = i % 3, r = i / 3, b = r / S;
  out[i] = xyz[((size_t)b * N + idx[r]) * 3 + c];
}

__global__ void k_ballquery(const float* xyz, const float* cen, int N, int S, int ns,
                            float r2, int* out, int total) {
  int i = blockIdx.x * blockDim.x + threadIdx.x;
  if (i >= total) return;
  int b = i / S;
  const float* c  = cen + (size_t)i * 3;
  const float* pb = xyz + (size_t)b * N * 3;
  int* o = out + (size_t)i * ns;
  int cnt = 0, first = 0; bool found = false;
  for (int j = 0; j < N; ++j) {
    float dx = pb[j * 3] - c[0], dy = pb[j * 3 + 1] - c[1], dz = pb[j * 3 + 2] - c[2];
    if (dx * dx + dy * dy + dz * dz < r2) {
      if (!found) { found = true; first = j; }
      o[cnt++] = j;
      if (cnt >= ns) break;
    }
  }
  int fill = found ? first : 0;
  for (int k = cnt; k < ns; ++k) o[k] = fill;
}

__global__ void k_pack_w(const float* W, _Float16* Wp, int co, int ci, int cop, int cip) {
  int i = blockIdx.x * blockDim.x + threadIdx.x;
  if (i >= cop * cip) return;
  int r = i / cip, c = i - r * cip;
  Wp[i] = (r < co && c < ci) ? (_Float16)W[r * ci + c] : (_Float16)0.0f;
}
__global__ void k_pack_b(const float* b, float* bp, int co, int cop) {
  int i = blockIdx.x * blockDim.x + threadIdx.x;
  if (i >= cop) return;
  bp[i] = (i < co) ? b[i] : 0.0f;
}

__global__ void k_cast16(const float* src, int ls, int cols, _Float16* dst, int ldd, int total) {
  int i = blockIdx.x * blockDim.x + threadIdx.x;
  if (i >= total) return;
  int r = i / ldd, c = i - r * ldd;
  dst[i] = (c < cols) ? (_Float16)src[(size_t)r * ls + c] : (_Float16)0.0f;
}

__global__ void k_concat16(const float* a, int la, int ca, const float* b, int lb, int cb,
                           _Float16* dst, int ldd, int total) {
  int i = blockIdx.x * blockDim.x + threadIdx.x;
  if (i >= total) return;
  int r = i / ldd, c = i - r * ldd;
  float v = 0.0f;
  if (c < ca)           v = a[(size_t)r * la + c];
  else if (c < ca + cb) v = b[(size_t)r * lb + (c - ca)];
  dst[i] = (_Float16)v;
}

// max / mean over the points of one batch (features stored [B, rows, ld])
__global__ void k_colstat(const float* src, int ld, int rows, int C,
                          float* omax, float* omean, int total) {
  int i = blockIdx.x * blockDim.x + threadIdx.x;
  if (i >= total) return;
  int b = i / C, c = i - b * C;
  float m = -3.0e38f, s = 0.0f;
  for (int r = 0; r < rows; ++r) {
    float v = src[((size_t)b * rows + r) * ld + c];
    m = fmaxf(m, v); s += v;
  }
  omax[i] = m;
  if (omean) omean[i] = s / rows;
}

__global__ void k_tga_in(const float* sf, int C, const float* tfp, _Float16* dst,
                         int Ss, int total) {
  int i = blockIdx.x * blockDim.x + threadIdx.x;
  if (i >= total) return;
  int ldd = 2 * C;
  int r = i / ldd, c = i - r * ldd;
  int b = r / Ss;
  float v = (c < C) ? sf[(size_t)r * C + c] : tfp[b * C + (c - C)];
  dst[i] = (_Float16)v;
}

__global__ void k_gatemul(const float* x, int ld, int C, const float* gate, int ldg,
                          float* out, int total) {
  int i = blockIdx.x * blockDim.x + threadIdx.x;
  if (i >= total) return;
  int r = i / C, c = i - r * C;
  out[(size_t)r * ld + c] = x[(size_t)r * ld + c] * gate[(size_t)r * ldg];
}

__global__ void k_copycol0(const float* src, int ld, float* dst, int n) {
  int i = blockIdx.x * blockDim.x + threadIdx.x;
  if (i < n) dst[i] = src[(size_t)i * ld];
}
__global__ void k_sigcol0(const float* src, int ld, float* dst, int n) {
  int i = blockIdx.x * blockDim.x + threadIdx.x;
  if (i < n) dst[i] = 1.0f / (1.0f + __expf(-src[(size_t)i * ld]));
}
__global__ void k_copyf(const float* s, float* d, int n) {
  int i = blockIdx.x * blockDim.x + threadIdx.x;
  if (i < n) d[i] = s[i];
}

// vote finish: v = fxf + raw; scatter xyz->d_out & vxc, feats(+score)->vfc
__global__ void k_votefin(const float* xyz, const float* fus, int ldf,
                          const float* raw, int ldr, const float* score,
                          float* vxyz_out, float* vxc, float* vfc, int FS,
                          int S, int poff, int total) {
  int i = blockIdx.x * blockDim.x + threadIdx.x;
  if (i >= total) return;
  int r = i / 259, c = i - r * 259;
  int b = r / S, n = r - b * S;
  float base = (c < 3) ? xyz[(size_t)r * 3 + c] : fus[(size_t)r * ldf + (c - 3)];
  float v = base + raw[(size_t)r * ldr + c];
  size_t pr = (size_t)b * 256 + poff + n;
  if (c < 3) { vxyz_out[(size_t)r * 3 + c] = v; vxc[pr * 3 + c] = v; }
  else       { vfc[pr * FS + 1 + (c - 3)] = v; }
  if (c == 0) vfc[pr * FS] = score[r];
}

__global__ void k_addsig(const float* a, const float* b, float* o, int n) {
  int i = blockIdx.x * blockDim.x + threadIdx.x;
  if (i < n) o[i] = 1.0f / (1.0f + __expf(-(a[i] + b[i])));
}
__global__ void k_cattmul(const float* x, int ld, const float* catt, int ptsPerB,
                          float* out, int total) {
  int i = blockIdx.x * blockDim.x + threadIdx.x;
  if (i >= total) return;
  int r = i / ld, c = i - r * ld;
  int b = r / ptsPerB;
  out[i] = x[i] * catt[b * ld + c];
}
__global__ void k_satt(const float* x, int C, const float* Wc, const float* bc,
                       float* satt, int rows) {
  int r = blockIdx.x * blockDim.x + threadIdx.x;
  if (r >= rows) return;
  float m = -3.0e38f, s = 0.0f;
  for (int c = 0; c < C; ++c) { float v = x[(size_t)r * C + c]; m = fmaxf(m, v); s += v; }
  float z = Wc[0] * m + Wc[1] * (s / C) + bc[0];
  satt[r] = 1.0f / (1.0f + __expf(-z));
}
__global__ void k_rowgate(const float* x, int ld, const float* g, float* out, int total) {
  int i = blockIdx.x * blockDim.x + threadIdx.x;
  if (i >= total) return;
  out[i] = x[i] * g[i / ld];
}
__global__ void k_est(const float* po, int ldo, const float* centers, float* dst, int total) {
  int i = blockIdx.x * blockDim.x + threadIdx.x;
  if (i >= total) return;
  int r = i / 5, c = i - r * 5;
  float v = po[(size_t)r * ldo + c];
  if (c < 3) v += centers[(size_t)r * 3 + c];
  dst[i] = v;
}

// ------------------------------------------------------------------- host side
namespace {

struct Arena {
  char* base; size_t off;
  void* get(size_t bytes) {
    off = (off + 255) & ~(size_t)255;
    void* p = base + off;
    off += bytes;
    return p;
  }
};

struct PLin { const _Float16* W; const float* b; int co, ci, cop, cip; };

inline int divup(int a, int b) { return (a + b - 1) / b; }

PLin pack_lin(hipStream_t st, Arena& a, void* const* d_in, int wi, int co, int ci) {
  PLin L; L.co = co; L.ci = ci;
  L.cop = (co + 15) & ~15; L.cip = (ci + 31) & ~31;
  _Float16* Wp = (_Float16*)a.get((size_t)L.cop * L.cip * sizeof(_Float16));
  float*    bp = (float*)a.get((size_t)L.cop * sizeof(float));
  int tot = L.cop * L.cip;
  k_pack_w<<<divup(tot, 256), 256, 0, st>>>((const float*)d_in[wi], Wp, co, ci, L.cop, L.cip);
  k_pack_b<<<divup(L.cop, 256), 256, 0, st>>>((const float*)d_in[wi + 1], bp, co, L.cop);
  L.W = Wp; L.b = bp;
  return L;
}

void gemm(hipStream_t st, const _Float16* A, int lda, const PLin& L, int M,
          int act, bool out16, void* out, int ldo) {
  int tm = M / 16;
  dim3 g(divup(tm, 8), L.cop / 16), blk(256);
  if (out16) {
    if (act == 1) k_gemm<1, 1><<<g, blk, 0, st>>>(A, lda, L.W, L.cip, L.b, out, ldo, tm, L.cip);
    else          k_gemm<0, 1><<<g, blk, 0, st>>>(A, lda, L.W, L.cip, L.b, out, ldo, tm, L.cip);
  } else {
    if (act == 2)      k_gemm<2, 0><<<g, blk, 0, st>>>(A, lda, L.W, L.cip, L.b, out, ldo, tm, L.cip);
    else if (act == 1) k_gemm<1, 0><<<g, blk, 0, st>>>(A, lda, L.W, L.cip, L.b, out, ldo, tm, L.cip);
    else               k_gemm<0, 0><<<g, blk, 0, st>>>(A, lda, L.W, L.cip, L.b, out, ldo, tm, L.cip);
  }
}

void run_sa(hipStream_t st, Arena scratch, const float* xyz, int N,
            const float* feats, int Cf, int FS, int S, int ns, float radius,
            const PLin* lins, int nl, float* out_xyz, float* out_f, int OS) {
  int* fi = (int*)scratch.get(sizeof(int) * (size_t)BATCH * S);
  k_fps<<<BATCH, 256, 0, st>>>(xyz, N, S, fi);
  int tot3 = BATCH * S * 3;
  k_gather3<<<divup(tot3, 256), 256, 0, st>>>(xyz, fi, out_xyz, N, S, tot3);
  int* bq = (int*)scratch.get(sizeof(int) * (size_t)BATCH * S * ns);
  int totc = BATCH * S;
  k_ballquery<<<divup(totc, 128), 128, 0, st>>>(xyz, out_xyz, N, S, ns,
                                               radius * radius, bq, totc);
  SAParams sp{};
  sp.xyz = xyz; sp.feats = feats; sp.idx = bq; sp.cen = out_xyz;
  sp.N = N; sp.S = S; sp.ns = ns; sp.Cf = Cf; sp.FS = FS;
  sp.nlayers = nl; sp.K0 = ((3 + Cf) + 31) & ~31;
  for (int i = 0; i < nl; ++i) { sp.Wl[i] = lins[i].W; sp.bl[i] = lins[i].b; sp.Co[i] = lins[i].cop; }
  sp.out = out_f; sp.OS = OS;
  k_sa<<<dim3(S, BATCH), 128, 0, st>>>(sp);
}

void run_backbone(hipStream_t st, Arena scratch, const float* pc, int N0,
                  int S1, int S2, int S3, const PLin* sa1, const PLin* sa2,
                  const PLin* sa3, const PLin& covf,
                  float* xyz2, float* xyz3, float* f2, float* f3) {
  float* xyz1 = (float*)scratch.get((size_t)BATCH * S1 * 3 * 4);
  float* f1   = (float*)scratch.get((size_t)BATCH * S1 * 128 * 4);
  run_sa(st, scratch, pc, N0, nullptr, 0, 0, S1, 32, 0.3f, sa1, 3, xyz1, f1, 128);
  run_sa(st, scratch, xyz1, S1, f1, 128, 128, S2, 32, 0.5f, sa2, 3, xyz2, f2, 256);
  float* f3p = (float*)scratch.get((size_t)BATCH * S3 * 256 * 4);
  run_sa(st, scratch, xyz2, S2, f2, 256, 256, S3, 32, 0.7f, sa3, 3, xyz3, f3p, 256);
  int M = BATCH * S3;
  _Float16* h = (_Float16*)scratch.get((size_t)M * 256 * sizeof(_Float16));
  k_cast16<<<divup(M * 256, 256), 256, 0, st>>>(f3p, 256, 256, h, 256, M * 256);
  gemm(st, h, 256, covf, M, 0, false, f3, 256);
}

void run_tga(hipStream_t st, Arena s, const float* sf, int Ss, const float* tf, int St,
             const PLin* L, float* out) {
  float* tfp = (float*)s.get((size_t)BATCH * 256 * 4);
  k_colstat<<<divup(BATCH * 256, 256), 256, 0, st>>>(tf, 256, St, 256, tfp, nullptr,
                                                    BATCH * 256);
  int M = BATCH * Ss;
  _Float16* x0 = (_Float16*)s.get((size_t)M * 512 * 2);
  k_tga_in<<<divup(M * 512, 256), 256, 0, st>>>(sf, 256, tfp, x0, Ss, M * 512);
  _Float16* x1 = (_Float16*)s.get((size_t)M * 256 * 2);
  gemm(st, x0, 512, L[0], M, 1, true, x1, 256);
  _Float16* x2 = (_Float16*)s.get((size_t)M * 128 * 2);
  gemm(st, x1, 256, L[1], M, 1, true, x2, 128);
  float* gate = (float*)s.get((size_t)M * 16 * 4);
  gemm(st, x2, 128, L[2], M, 2, false, gate, 16);
  k_gatemul<<<divup(M * 256, 256), 256, 0, st>>>(sf, 256, 256, gate, 16, out, M * 256);
}

void run_cla(hipStream_t st, Arena s, const float* feat, int S, const PLin* L,
             float* cla_out, float* score) {
  int M = BATCH * S;
  _Float16* x0 = (_Float16*)s.get((size_t)M * 256 * 2);
  k_cast16<<<divup(M * 256, 256), 256, 0, st>>>(feat, 256, 256, x0, 256, M * 256);
  _Float16* x1 = (_Float16*)s.get((size_t)M * 256 * 2);
  gemm(st, x0, 256, L[0], M, 1, true, x1, 256);
  _Float16* x2 = (_Float16*)s.get((size_t)M * 256 * 2);
  gemm(st, x1, 256, L[1], M, 1, true, x2, 256);
  float* raw = (float*)s.get((size_t)M * 16 * 4);
  gemm(st, x2, 256, L[2], M, 0, false, raw, 16);
  k_copycol0<<<divup(M, 256), 256, 0, st>>>(raw, 16, cla_out, M);
  k_sigcol0<<<divup(M, 256), 256, 0, st>>>(raw, 16, score, M);
}

void run_vote(hipStream_t st, Arena s, const float* xyz, const float* feat, int S,
              const PLin* L, const float* score, float* vxyz_out,
              float* vxc, float* vfc, int FS, int poff) {
  int M = BATCH * S;
  _Float16* x0 = (_Float16*)s.get((size_t)M * 288 * 2);
  k_concat16<<<divup(M * 288, 256), 256, 0, st>>>(xyz, 3, 3, feat, 256, 256, x0, 288,
                                                  M * 288);
  _Float16* x1 = (_Float16*)s.get((size_t)M * 256 * 2);
  gemm(st, x0, 288, L[0], M, 1, true, x1, 256);
  _Float16* x2 = (_Float16*)s.get((size_t)M * 256 * 2);
  gemm(st, x1, 256, L[1], M, 1, true, x2, 256);
  float* raw = (float*)s.get((size_t)M * 272 * 4);
  gemm(st, x2, 256, L[2], M, 0, false, raw, 272);
  k_votefin<<<divup(M * 259, 256), 256, 0, st>>>(xyz, feat, 256, raw, 272, score,
                                                 vxyz_out, vxc, vfc, FS, S, poff,
                                                 M * 259);
}

void run_cbam_prop(hipStream_t st, Arena s, const float* pf, const float* centers,
                   const float* cbW, const float* cbB, const PLin* m1, const PLin* pr,
                   float* est_out) {
  float* mp = (float*)s.get((size_t)BATCH * 256 * 4);
  float* ap = (float*)s.get((size_t)BATCH * 256 * 4);
  k_colstat<<<divup(BATCH * 256, 256), 256, 0, st>>>(pf, 256, 64, 256, mp, ap,
                                                    BATCH * 256);
  _Float16* mph = (_Float16*)s.get((size_t)BATCH * 256 * 2);
  _Float16* aph = (_Float16*)s.get((size_t)BATCH * 256 * 2);
  k_cast16<<<divup(BATCH * 256, 256), 256, 0, st>>>(mp, 256, 256, mph, 256, BATCH * 256);
  k_cast16<<<divup(BATCH * 256, 256), 256, 0, st>>>(ap, 256, 256, aph, 256, BATCH * 256);
  _Float16* t1  = (_Float16*)s.get((size_t)BATCH * 256 * 2);
  float*    yma = (float*)s.get((size_t)BATCH * 256 * 4);
  float*    yaa = (float*)s.get((size_t)BATCH * 256 * 4);
  gemm(st, mph, 256, m1[0], BATCH, 1, true, t1, 256);
  gemm(st, t1, 256, m1[1], BATCH, 0, false, yma, 256);
  gemm(st, aph, 256, m1[0], BATCH, 1, true, t1, 256);
  gemm(st, t1, 256, m1[1], BATCH, 0, false, yaa, 256);
  float* catt = (float*)s.get((size_t)BATCH * 256 * 4);
  k_addsig<<<divup(BATCH * 256, 256), 256, 0, st>>>(yma, yaa, catt, BATCH * 256);
  int M = BATCH * 64;
  float* x1 = (float*)s.get((size_t)M * 256 * 4);
  k_cattmul<<<divup(M * 256, 256), 256, 0, st>>>(pf, 256, catt, 64, x1, M * 256);
  float* satt = (float*)s.get((size_t)M * 4);
  k_satt<<<divup(M, 64), 64, 0, st>>>(x1, 256, cbW, cbB, satt, M);
  float* pf2 = (float*)s.get((size_t)M * 256 * 4);
  k_rowgate<<<divup(M * 256, 256), 256, 0, st>>>(x1, 256, satt, pf2, M * 256);
  // proposal head
  _Float16* p0 = (_Float16*)s.get((size_t)M * 256 * 2);
  k_cast16<<<divup(M * 256, 256), 256, 0, st>>>(pf2, 256, 256, p0, 256, M * 256);
  _Float16* p1 = (_Float16*)s.get((size_t)M * 256 * 2);
  gemm(st, p0, 256, pr[0], M, 1, true, p1, 256);
  _Float16* p2 = (_Float16*)s.get((size_t)M * 256 * 2);
  gemm(st, p1, 256, pr[1], M, 1, true, p2, 256);
  float* po = (float*)s.get((size_t)M * 16 * 4);
  gemm(st, p2, 256, pr[2], M, 0, false, po, 16);
  k_est<<<divup(M * 5, 256), 256, 0, st>>>(po, 16, centers, est_out, M * 5);
}

}  // namespace

extern "C" void kernel_launch(void* const* d_in, const int* in_sizes, int n_in,
                              void* d_out, int out_size, void* d_ws, size_t ws_size,
                              hipStream_t stream) {
  (void)in_sizes; (void)n_in; (void)out_size; (void)ws_size;
  const float* tmpl = (const float*)d_in[0];   // [64,512,3]
  const float* srch = (const float*)d_in[1];   // [64,1024,3]
  float* out = (float*)d_out;
  Arena a{(char*)d_ws, 0};

  // --- pack weights (jax pytree flatten order: sorted dict keys, 'W' < 'b') ---
  PLin agg[3]  = {pack_lin(stream, a, d_in, 2, 256, 260),
                  pack_lin(stream, a, d_in, 4, 256, 256),
                  pack_lin(stream, a, d_in, 6, 256, 256)};
  PLin covf    =  pack_lin(stream, a, d_in, 8, 256, 256);
  PLin sa1[3]  = {pack_lin(stream, a, d_in, 10, 64, 3),
                  pack_lin(stream, a, d_in, 12, 64, 64),
                  pack_lin(stream, a, d_in, 14, 128, 64)};
  PLin sa2[3]  = {pack_lin(stream, a, d_in, 16, 128, 131),
                  pack_lin(stream, a, d_in, 18, 128, 128),
                  pack_lin(stream, a, d_in, 20, 256, 128)};
  PLin sa3[3]  = {pack_lin(stream, a, d_in, 22, 256, 259),
                  pack_lin(stream, a, d_in, 24, 256, 256),
                  pack_lin(stream, a, d_in, 26, 256, 256)};
  const float* cbW = (const float*)d_in[28];   // cbam.conv.W (1,2)
  const float* cbB = (const float*)d_in[29];
  PLin m1[2]   = {pack_lin(stream, a, d_in, 30, 256, 256),
                  pack_lin(stream, a, d_in, 32, 256, 256)};
  PLin claL[3] = {pack_lin(stream, a, d_in, 34, 256, 256),
                  pack_lin(stream, a, d_in, 36, 256, 256),
                  pack_lin(stream, a, d_in, 38, 1, 256)};
  PLin cla2L[3]= {pack_lin(stream, a, d_in, 40, 256, 256),
                  pack_lin(stream, a, d_in, 42, 256, 256),
                  pack_lin(stream, a, d_in, 44, 1, 256)};
  PLin prop[3] = {pack_lin(stream, a, d_in, 46, 256, 256),
                  pack_lin(stream, a, d_in, 48, 256, 256),
                  pack_lin(stream, a, d_in, 50, 5, 256)};
  PLin tg2[3]  = {pack_lin(stream, a, d_in, 52, 256, 512),
                  pack_lin(stream, a, d_in, 54, 128, 256),
                  pack_lin(stream, a, d_in, 56, 1, 128)};
  PLin tg3[3]  = {pack_lin(stream, a, d_in, 58, 256, 512),
                  pack_lin(stream, a, d_in, 60, 128, 256),
                  pack_lin(stream, a, d_in, 62, 1, 128)};
  PLin tr2[1]  = {pack_lin(stream, a, d_in, 64, 256, 259)};
  PLin vt[3]   = {pack_lin(stream, a, d_in, 66, 256, 259),
                  pack_lin(stream, a, d_in, 68, 256, 256),
                  pack_lin(stream, a, d_in, 70, 259, 256)};
  PLin vt2[3]  = {pack_lin(stream, a, d_in, 72, 256, 259),
                  pack_lin(stream, a, d_in, 74, 256, 256),
                  pack_lin(stream, a, d_in, 76, 259, 256)};

  // --- persistent activation buffers (point-major [B, N, C]) ---
  float* t_xyz2 = (float*)a.get((size_t)BATCH * 128 * 3 * 4);
  float* t_xyz3 = (float*)a.get((size_t)BATCH * 64 * 3 * 4);
  float* t_f2   = (float*)a.get((size_t)BATCH * 128 * 256 * 4);
  float* t_f3   = (float*)a.get((size_t)BATCH * 64 * 256 * 4);
  float* s_xyz2 = (float*)a.get((size_t)BATCH * 256 * 3 * 4);
  float* s_xyz3 = (float*)a.get((size_t)BATCH * 128 * 3 * 4);
  float* s_f2   = (float*)a.get((size_t)BATCH * 256 * 256 * 4);
  float* s_f3   = (float*)a.get((size_t)BATCH * 128 * 256 * 4);
  float* nsf2   = (float*)a.get((size_t)BATCH * 256 * 256 * 4);
  float* fusion = (float*)a.get((size_t)BATCH * 128 * 256 * 4);
  float* v2_xyz = (float*)a.get((size_t)BATCH * 128 * 3 * 4);
  float* v2_f   = (float*)a.get((size_t)BATCH * 128 * 256 * 4);
  float* score2 = (float*)a.get((size_t)BATCH * 128 * 4);
  float* score  = (float*)a.get((size_t)BATCH * 128 * 4);
  float* vxc    = (float*)a.get((size_t)BATCH * 256 * 3 * 4);
  float* vfc    = (float*)a.get((size_t)BATCH * 256 * 257 * 4);
  float* cent   = (float*)a.get((size_t)BATCH * 64 * 3 * 4);
  float* pf     = (float*)a.get((size_t)BATCH * 64 * 256 * 4);

  // d_out layout (floats): cla2, cla, vote_xyz2, vote_xyz, est, centers
  const size_t o_cla2 = 0, o_cla = 8192, o_vx2 = 16384, o_vx = 40960,
               o_est = 65536, o_cen = 86016;

  // --- backbones (scratch passed by value: transient regions reuse the tail) ---
  run_backbone(stream, a, tmpl, 512, 256, 128, 64, sa1, sa2, sa3, covf,
               t_xyz2, t_xyz3, t_f2, t_f3);
  run_backbone(stream, a, srch, 1024, 512, 256, 128, sa1, sa2, sa3, covf,
               s_xyz2, s_xyz3, s_f2, s_f3);

  // --- TGA fusion ---
  run_tga(stream, a, s_f2, 256, t_f2, 128, tg2, nsf2);
  run_tga(stream, a, s_f3, 128, t_f3, 64, tg3, fusion);

  // --- trans2 SA ---
  run_sa(stream, a, s_xyz2, 256, nsf2, 256, 256, 128, 32, 0.3f, tr2, 1,
         v2_xyz, v2_f, 256);

  // --- classification + vote heads ---
  run_cla(stream, a, v2_f, 128, cla2L, out + o_cla2, score2);
  run_vote(stream, a, v2_xyz, v2_f, 128, vt2, score2, out + o_vx2, vxc, vfc, 257, 0);
  run_cla(stream, a, fusion, 128, claL, out + o_cla, score);
  run_vote(stream, a, s_xyz3, fusion, 128, vt, score, out + o_vx, vxc, vfc, 257, 128);

  // --- aggregation SA ---
  run_sa(stream, a, vxc, 256, vfc, 257, 257, 64, 48, 0.3f, agg, 3, cent, pf, 256);
  k_copyf<<<divup(BATCH * 64 * 3, 256), 256, 0, stream>>>(cent, out + o_cen,
                                                          BATCH * 64 * 3);

  // --- CBAM + proposal ---
  run_cbam_prop(stream, a, pf, cent, cbW, cbB, m1, prop, out + o_est);
}
